// UltralightMamba_52570399703635
// MI455X (gfx1250) — compile-verified, matrix operand order checked by reading, not run on hardware
//
#include <hip/hip_runtime.h>
#include <hip/hip_bf16.h>
#include <stdint.h>

// ---------------------------------------------------------------------------
// CDNA5 (gfx1250) fused Mamba block.
// - GEMMs on v_wmma_f32_16x16x32_bf16 (weight fragments VGPR-resident)
// - selective scan state in registers (one channel per thread, whole sequence)
// - double-buffered activation tiles via global_load_async_to_lds_b128
// ---------------------------------------------------------------------------

typedef __attribute__((ext_vector_type(16))) __bf16 v16bf;
typedef __attribute__((ext_vector_type(8)))  float  v8f;

union FragU { uint4 q[2]; v16bf v; };

__device__ __forceinline__ unsigned short f2bf(float f) {
  union { float f; unsigned u; } cv; cv.f = f;
  unsigned u = cv.u;
  u += 0x7FFFu + ((u >> 16) & 1u);          // round-to-nearest-even
  return (unsigned short)(u >> 16);
}

__device__ __forceinline__ v8f vzero8() {
  v8f z = {0.f,0.f,0.f,0.f,0.f,0.f,0.f,0.f};
  return z;
}

__device__ __forceinline__ v8f wmma_bf16(v16bf a, v16bf b, v8f c) {
  // 8 args: (neg_a, A, neg_b, B, c_mod, C, reuse_a, reuse_b)
  return __builtin_amdgcn_wmma_f32_16x16x32_bf16(false, a, false, b, (short)0, c,
                                                 false, false);
}

// 16x32 bf16 A/B fragment from LDS (row-major, ld in elements).
// ISA 7.12.2: lane<16 holds row=lane, K in {0..7,16..23}; lane>=16 holds
// row=lane-16, K in {8..15,24..31} -> two contiguous 16B loads per lane.
__device__ __forceinline__ v16bf frag_lds(const unsigned short* s, int row0,
                                          int k0, int ld, int lane) {
  const int lm = lane & 15;
  const int b8 = (lane >> 4) << 3;
  const unsigned short* p = s + (size_t)(row0 + lm) * ld + k0 + b8;
  FragU f;
  f.q[0] = *(const uint4*)(p);
  f.q[1] = *(const uint4*)(p + 16);
  return f.v;
}

// Same fragment from global bf16 weights, with row bounds (zero-pad).
__device__ __forceinline__ v16bf frag_gbl(const unsigned short* g, int row0,
                                          int k0, int ld, int lane, int nrows) {
  const int lm = lane & 15;
  const int b8 = (lane >> 4) << 3;
  const int r  = row0 + lm;
  FragU f;
  if (r < nrows) {
    const unsigned short* p = g + (size_t)r * ld + k0 + b8;
    f.q[0] = *(const uint4*)(p);
    f.q[1] = *(const uint4*)(p + 16);
  } else {
    f.q[0] = make_uint4(0u,0u,0u,0u);
    f.q[1] = make_uint4(0u,0u,0u,0u);
  }
  return f.v;
}

// 16x16 f32 C/D fragment -> LDS (VGPR r, lane l => row r+8*(l>=16), col l&15).
__device__ __forceinline__ void store_frag(float* s, v8f c, int m0, int n0,
                                           int ld, int lane) {
  const int ln = lane & 15;
  const int mo = (lane >> 4) << 3;
  #pragma unroll
  for (int r = 0; r < 8; ++r)
    s[(size_t)(m0 + mo + r) * ld + n0 + ln] = c[r];
}

// Async copy 16 bytes global -> LDS (ASYNCcnt-tracked, CDNA5 path).
__device__ __forceinline__ void async_cp16(unsigned lds_off,
                                           const unsigned short* gptr) {
  asm volatile("global_load_async_to_lds_b128 %0, %1, off"
               :: "v"(lds_off), "v"(gptr) : "memory");
}
__device__ __forceinline__ void wait_async0() {
  asm volatile("s_wait_asynccnt 0" ::: "memory");
}

// ---------------------------------------------------------------------------
// Kernel 0: f32 -> bf16 weight conversion (weights then live in L2).
// ---------------------------------------------------------------------------
__global__ __launch_bounds__(256) void k_cvt_bf16(const float* __restrict__ src,
                                                  unsigned short* __restrict__ dst,
                                                  int n) {
  const int i = blockIdx.x * 256 + threadIdx.x;
  if (i < n) dst[i] = f2bf(src[i]);
}

// ---------------------------------------------------------------------------
// Kernel 1: layernorm over C=512 + split into 4 chunks of 128.
// Emits bf16 (GEMM operand) and f32 (skip path) in [s][l][128] layout,
// s = chunk*64 + b*32 + h.  One wave32 per token row.
// ---------------------------------------------------------------------------
__global__ __launch_bounds__(256) void k_ln_split(const float* __restrict__ x,
                                                  const float* __restrict__ lw,
                                                  const float* __restrict__ lb,
                                                  unsigned short* __restrict__ xb,
                                                  float* __restrict__ xf) {
  const int lane = threadIdx.x & 31;
  const int wv   = threadIdx.x >> 5;
  const int t  = blockIdx.x * 8 + wv;          // token 0..32767
  const int b  = t >> 14;
  const int h  = (t >> 9) & 31;
  const int dd = t & 511;
  const float* row = x + (size_t)t * 512;

  float v[16];
  float sum = 0.f, sq = 0.f;
  #pragma unroll
  for (int i = 0; i < 16; ++i) {
    const float a = row[lane + (i << 5)];
    v[i] = a; sum += a; sq += a * a;
  }
  #pragma unroll
  for (int d = 16; d > 0; d >>= 1) {
    sum += __shfl_xor(sum, d, 32);
    sq  += __shfl_xor(sq,  d, 32);
  }
  const float m  = sum * (1.f / 512.f);
  const float rs = rsqrtf(sq * (1.f / 512.f) - m * m + 1e-5f);

  #pragma unroll
  for (int i = 0; i < 16; ++i) {
    const int c  = lane + (i << 5);
    const float xn = (v[i] - m) * rs * lw[c] + lb[c];
    const int cc = c >> 7, j = c & 127;
    const size_t off = ((size_t)((cc * 2 + b) * 32 + h) * 512 + dd) * 128 + j;
    xb[off] = f2bf(xn);
    xf[off] = xn;
  }
}

// ---------------------------------------------------------------------------
// Kernel 2: fused Mamba inner block.  One block (8 waves) owns one sequence
// (512 tokens); iterates 32 tiles of 16 tokens.  Thread t == channel d.
// Activation tiles are double-buffered with async global->LDS DMA.
// ---------------------------------------------------------------------------
__global__ __launch_bounds__(256) void k_mamba(
    const unsigned short* __restrict__ xb,   // [256][512][128] bf16
    const float* __restrict__ xf,            // [256][512][128] f32 (skip)
    const unsigned short* __restrict__ wip,  // in_proj  [512][128] bf16
    const float* __restrict__ conv_w,        // [256][4]
    const float* __restrict__ conv_b,        // [256]
    const unsigned short* __restrict__ wxp,  // x_proj   [40][256]  bf16
    const float* __restrict__ dtw_g,         // dt_proj_w [256][8]
    const float* __restrict__ dtb_g,         // dt_proj_b [256]
    const float* __restrict__ A_log,         // [256][16]
    const float* __restrict__ Dp_g,          // [256]
    const unsigned short* __restrict__ wop,  // out_proj [128][256] bf16
    const float* __restrict__ skip_g,        // [1]
    float* __restrict__ ym)                  // [256][512][128]
{
  __shared__ alignas(16) unsigned short sXb[2][16 * 128]; // x tiles bf16 (8 KB)
  __shared__ alignas(16) float          sXI[16 * 256];    // conv input  (16 KB)
  __shared__ alignas(16) float          sZ [16 * 256];    // gate input  (16 KB)
  __shared__ alignas(16) unsigned short sXS[16 * 256];    // xs / gated   (8 KB)
  __shared__ alignas(16) float          sDBL[16 * 48];    // dt|B|C pad48 (3 KB)

  const int tid  = threadIdx.x;
  const int lane = tid & 31;
  const int wv   = tid >> 5;
  const int s    = blockIdx.x;
  const size_t seq_base = (size_t)s * (512 * 128);

  // ---- per-channel constants in registers -------------------------------
  const float cw0 = conv_w[tid * 4 + 0], cw1 = conv_w[tid * 4 + 1];
  const float cw2 = conv_w[tid * 4 + 2], cw3 = conv_w[tid * 4 + 3];
  const float cbv  = conv_b[tid];
  const float dtb  = dtb_g[tid];
  const float Dpv  = Dp_g[tid];
  const float skip = skip_g[0];
  float dtw[8];
  #pragma unroll
  for (int r = 0; r < 8; ++r) dtw[r] = dtw_g[tid * 8 + r];
  float Av[16];
  #pragma unroll
  for (int n = 0; n < 16; ++n) Av[n] = -__expf(A_log[tid * 16 + n]);

  // ---- hoisted weight fragments: loaded once, reused 32x ---------------
  v16bf bip[4][4];                        // in_proj: 4 N-tiles x 4 K-steps
  #pragma unroll
  for (int t = 0; t < 4; ++t)
    #pragma unroll
    for (int kk = 0; kk < 4; ++kk)
      bip[t][kk] = frag_gbl(wip, (wv + t * 8) * 16, kk * 32, 128, lane, 512);
  v16bf bop[8];                           // out_proj: N-tile = wave
  #pragma unroll
  for (int kk = 0; kk < 8; ++kk)
    bop[kk] = frag_gbl(wop, wv * 16, kk * 32, 256, lane, 128);
  v16bf bxp[8];                           // x_proj: waves 0..2 (rows>=40 zero)
  if (wv < 3) {
    #pragma unroll
    for (int kk = 0; kk < 8; ++kk)
      bxp[kk] = frag_gbl(wxp, wv * 16, kk * 32, 256, lane, 40);
  }

  // ---- persistent per-channel recurrent state --------------------------
  float hst[16];
  #pragma unroll
  for (int n = 0; n < 16; ++n) hst[n] = 0.f;
  float c0 = 0.f, c1 = 0.f, c2 = 0.f;     // conv taps x[l-3], x[l-2], x[l-1]

  // ---- kick off async DMA of tile 0 into buffer 0 ----------------------
  const unsigned sxb_lds = (unsigned)(size_t)(void*)&sXb[0][0];
  async_cp16(sxb_lds + tid * 16, xb + seq_base + tid * 8);

  #pragma unroll 1
  for (int it = 0; it < 32; ++it) {
    const int l0  = it * 16;
    const int buf = it & 1;

    // stage 1: publish current tile; prefetch next tile asynchronously
    wait_async0();            // my 16B slice of tile `it` has landed in LDS
    __syncthreads();          // everyone's slices have landed
    if (it + 1 < 32)
      async_cp16(sxb_lds + (buf ^ 1) * 4096 + tid * 16,
                 xb + seq_base + (size_t)(l0 + 16) * 128 + tid * 8);

    // stage 2: in_proj GEMM (16x128)@(128->512), split xi | z
    {
      const unsigned short* sx = &sXb[buf][0];
      v16bf afr[4];
      #pragma unroll
      for (int kk = 0; kk < 4; ++kk)
        afr[kk] = frag_lds(sx, 0, kk * 32, 128, lane);
      #pragma unroll
      for (int t = 0; t < 4; ++t) {
        v8f acc = vzero8();
        #pragma unroll
        for (int kk = 0; kk < 4; ++kk)
          acc = wmma_bf16(afr[kk], bip[t][kk], acc);
        const int nt = wv + t * 8;
        if (nt < 16) store_frag(sXI, acc, 0, nt * 16, 256, lane);
        else         store_frag(sZ,  acc, 0, (nt - 16) * 16, 256, lane);
      }
    }
    __syncthreads();

    // stage 3: depthwise causal conv(4) + SiLU, history in registers
    float xsl[16];
    #pragma unroll
    for (int l = 0; l < 16; ++l) {
      const float cur = sXI[l * 256 + tid];
      const float a = cw0 * c0 + cw1 * c1 + cw2 * c2 + cw3 * cur + cbv;
      c0 = c1; c1 = c2; c2 = cur;
      const float xs = a / (1.f + __expf(-a));
      xsl[l] = xs;
      sXS[l * 256 + tid] = f2bf(xs);
    }
    __syncthreads();

    // stage 4: x_proj GEMM (16x256)@(256->40, padded to 48)
    if (wv < 3) {
      v16bf afr[8];
      #pragma unroll
      for (int kk = 0; kk < 8; ++kk)
        afr[kk] = frag_lds(sXS, 0, kk * 32, 256, lane);
      v8f acc = vzero8();
      #pragma unroll
      for (int kk = 0; kk < 8; ++kk)
        acc = wmma_bf16(afr[kk], bxp[kk], acc);
      store_frag(sDBL, acc, 0, wv * 16, 48, lane);
    }
    __syncthreads();

    // stage 5: dt (softplus) + selective scan + SiLU gate, per channel
    #pragma unroll
    for (int l = 0; l < 16; ++l) {
      float dtv = dtb;
      #pragma unroll
      for (int r = 0; r < 8; ++r) dtv += sDBL[l * 48 + r] * dtw[r];
      dtv = (dtv > 20.f) ? dtv : __logf(1.f + __expf(dtv));
      const float xv = xsl[l];
      const float dx = dtv * xv;
      float y = 0.f;
      #pragma unroll
      for (int n = 0; n < 16; ++n) {
        hst[n] = __expf(Av[n] * dtv) * hst[n] + dx * sDBL[l * 48 + 8 + n];
        y += hst[n] * sDBL[l * 48 + 24 + n];
      }
      const float zv = sZ[l * 256 + tid];
      const float g  = (y + Dpv * xv) * (zv / (1.f + __expf(-zv)));
      sXS[l * 256 + tid] = f2bf(g);     // reuse buffer: gated activation
    }
    __syncthreads();

    // stage 6: out_proj GEMM (16x256)@(256->128) + skip, store ym
    {
      v16bf afr[8];
      #pragma unroll
      for (int kk = 0; kk < 8; ++kk)
        afr[kk] = frag_lds(sXS, 0, kk * 32, 256, lane);
      v8f acc = vzero8();
      #pragma unroll
      for (int kk = 0; kk < 8; ++kk)
        acc = wmma_bf16(afr[kk], bop[kk], acc);
      const int ln = lane & 15;
      const int mo = (lane >> 4) << 3;
      const size_t base = seq_base + (size_t)l0 * 128 + wv * 16 + ln;
      #pragma unroll
      for (int r = 0; r < 8; ++r) {
        const size_t idx = base + (size_t)(mo + r) * 128;
        ym[idx] = acc[r] + skip * xf[idx];
      }
    }
    __syncthreads();
  }
}

// ---------------------------------------------------------------------------
// Kernel 3: de-chunk gather + layernorm + final (32768x512)@(512x512) GEMM.
// One block = 32 token rows; each proj_w fragment feeds BOTH m-tiles.
// ---------------------------------------------------------------------------
__global__ __launch_bounds__(256) void k_final(const float* __restrict__ ym,
                                               const float* __restrict__ lw,
                                               const float* __restrict__ lb,
                                               const unsigned short* __restrict__ wpr,
                                               const float* __restrict__ pb,
                                               float* __restrict__ out) {
  __shared__ alignas(16) unsigned short sR[32 * 512];   // 32 KB

  const int tid = threadIdx.x, lane = tid & 31, wv = tid >> 5;
  const int m0 = blockIdx.x * 32;

  #pragma unroll 1
  for (int rr = 0; rr < 4; ++rr) {
    const int row = wv * 4 + rr;
    const int t = m0 + row;
    const int b = t >> 14, h = (t >> 9) & 31, dd = t & 511;
    float v[16];
    float sum = 0.f, sq = 0.f;
    #pragma unroll
    for (int i = 0; i < 16; ++i) {
      const int c = lane + (i << 5);
      const int cc = c >> 7, j = c & 127;
      const float a = ym[((size_t)((cc * 2 + b) * 32 + h) * 512 + dd) * 128 + j];
      v[i] = a; sum += a; sq += a * a;
    }
    #pragma unroll
    for (int d = 16; d > 0; d >>= 1) {
      sum += __shfl_xor(sum, d, 32);
      sq  += __shfl_xor(sq,  d, 32);
    }
    const float m  = sum * (1.f / 512.f);
    const float rs = rsqrtf(sq * (1.f / 512.f) - m * m + 1e-5f);
    #pragma unroll
    for (int i = 0; i < 16; ++i) {
      const int c = lane + (i << 5);
      sR[row * 512 + c] = f2bf((v[i] - m) * rs * lw[c] + lb[c]);
    }
  }
  __syncthreads();

  // A fragments for both 16-row m-tiles stay resident; each B fragment is
  // loaded once and consumed by two independent WMMAs (latency overlap).
  v16bf afr[2][16];
  #pragma unroll
  for (int mt = 0; mt < 2; ++mt)
    #pragma unroll
    for (int kk = 0; kk < 16; ++kk)
      afr[mt][kk] = frag_lds(sR, mt * 16, kk * 32, 512, lane);

  #pragma unroll 1
  for (int t = 0; t < 4; ++t) {
    const int n0 = (wv + t * 8) * 16;
    v8f acc0 = vzero8();
    v8f acc1 = vzero8();
    #pragma unroll
    for (int kk = 0; kk < 16; ++kk) {
      const v16bf bfr = frag_gbl(wpr, n0, kk * 32, 512, lane, 512);
      acc0 = wmma_bf16(afr[0][kk], bfr, acc0);
      acc1 = wmma_bf16(afr[1][kk], bfr, acc1);
    }
    const int ln = lane & 15;
    const int mo = (lane >> 4) << 3;
    const int col = n0 + ln;
    const float bias = pb[col];
    #pragma unroll
    for (int r = 0; r < 8; ++r) {
      const int row = m0 + mo + r;
      out[(size_t)row * 512 + col]        = acc0[r] + bias;
      out[(size_t)(row + 16) * 512 + col] = acc1[r] + bias;
    }
  }
}

// ---------------------------------------------------------------------------
extern "C" void kernel_launch(void* const* d_in, const int* in_sizes, int n_in,
                              void* d_out, int out_size, void* d_ws, size_t ws_size,
                              hipStream_t stream) {
  (void)in_sizes; (void)n_in; (void)out_size; (void)ws_size;
  const float* x     = (const float*)d_in[0];
  const float* ln_w  = (const float*)d_in[1];
  const float* ln_b  = (const float*)d_in[2];
  const float* skip  = (const float*)d_in[3];
  const float* wip_f = (const float*)d_in[4];
  const float* convw = (const float*)d_in[5];
  const float* convb = (const float*)d_in[6];
  const float* wxp_f = (const float*)d_in[7];
  const float* dtw   = (const float*)d_in[8];
  const float* dtb   = (const float*)d_in[9];
  const float* alog  = (const float*)d_in[10];
  const float* dpar  = (const float*)d_in[11];
  const float* wop_f = (const float*)d_in[12];
  const float* wpr_f = (const float*)d_in[13];
  const float* projb = (const float*)d_in[14];
  float* out = (float*)d_out;

  char* ws = (char*)d_ws;
  size_t o = 0;
  auto alloc = [&](size_t bytes) -> char* {
    char* p = ws + o;
    o = (o + bytes + 255) & ~(size_t)255;
    return p;
  };
  unsigned short* xb  = (unsigned short*)alloc((size_t)256 * 512 * 128 * 2);
  float*          xf  = (float*)         alloc((size_t)256 * 512 * 128 * 4);
  float*          ymw = (float*)         alloc((size_t)256 * 512 * 128 * 4);
  unsigned short* wip = (unsigned short*)alloc((size_t)512 * 128 * 2);
  unsigned short* wxp = (unsigned short*)alloc((size_t)40 * 256 * 2);
  unsigned short* wop = (unsigned short*)alloc((size_t)128 * 256 * 2);
  unsigned short* wpr = (unsigned short*)alloc((size_t)512 * 512 * 2);

  k_cvt_bf16<<<(512 * 128 + 255) / 256, 256, 0, stream>>>(wip_f, wip, 512 * 128);
  k_cvt_bf16<<<(40 * 256 + 255) / 256, 256, 0, stream>>>(wxp_f, wxp, 40 * 256);
  k_cvt_bf16<<<(128 * 256 + 255) / 256, 256, 0, stream>>>(wop_f, wop, 128 * 256);
  k_cvt_bf16<<<(512 * 512 + 255) / 256, 256, 0, stream>>>(wpr_f, wpr, 512 * 512);

  k_ln_split<<<4096, 256, 0, stream>>>(x, ln_w, ln_b, xb, xf);
  k_mamba<<<256, 256, 0, stream>>>(xb, xf, wip, convw, convb, wxp, dtw, dtb,
                                   alog, dpar, wop, skip, ymw);
  k_final<<<1024, 256, 0, stream>>>(ymw, ln_w, ln_b, wpr, projb, out);
}
